// E3nnEdgeMessageBlock_75471165325415
// MI455X (gfx1250) — matrix operand order repaired and checked by reading, not compiled, not run on hardware
//
#include <hip/hip_runtime.h>
#include <hip/hip_bf16.h>

typedef _Float16 half_t;
typedef __attribute__((ext_vector_type(16))) _Float16 v16h;
typedef __attribute__((ext_vector_type(8)))  _Float16 v8h;
typedef __attribute__((ext_vector_type(8)))  float    v8f;

#define DEVINL __device__ __forceinline__

static constexpr int N_NODES = 50000;
static constexpr int N_EDGES = 400000;
static constexpr float INV_SQRT3 = 0.57735026918962576451f;

// ---- packed B-fragment directory (frag = 32 lanes x 16 halves = 1024 B) ----
static constexpr int FRAG_UP0 = 0;    // 8:  nt(0..3)*2 + kc(0..1)
static constexpr int FRAG_UP1 = 8;    // 8
static constexpr int FRAG_MW1 = 16;   // 4:  nt(0..3)          (K=8 padded)
static constexpr int FRAG_MW2 = 20;   // 8:  nt*2 + kc
static constexpr int FRAG_MW3 = 28;   // 8
static constexpr int FRAG_MW4 = 36;   // 32: nt(0..15)*2 + kc
static constexpr int FRAG_WL0 = 68;   // 16: nt(0..3)*4 + kc(0..3)
static constexpr int FRAG_WL1 = 84;   // 16
static constexpr int N_FRAGS  = 100;
static constexpr size_t PK_BYTES = (size_t)N_FRAGS * 32 * 16 * sizeof(half_t); // 102400

// ---------------------------------------------------------------------------
DEVINL v8f wmma_f16(v16h a, v16h b, v8f c) {
  return __builtin_amdgcn_wmma_f32_16x16x32_f16(
      false, a, false, b, (short)0, c, false, false);
}

// B fragment built from row-major fp32 weight (pack kernel only).
// Lane layout: n = n0+(lane&15); kb = k0 + 8*(lane>=16); half i -> K = kb+i (i<8), kb+8+i (i>=8).
DEVINL v16h make_b_frag(const float* __restrict__ W, int ldN, int k0, int n0,
                        int kmax, int lane) {
  const int n  = n0 + (lane & 15);
  const int kb = k0 + ((lane & 16) ? 8 : 0);
  v16h b;
#pragma unroll
  for (int i = 0; i < 16; ++i) {
    const int K = kb + (i < 8 ? i : i + 8);
    b[i] = (K < kmax) ? (half_t)W[K * ldN + n] : (half_t)0.0f;
  }
  return b;
}

// Packed B fragment: two coalesced 16-byte loads per lane.
DEVINL v16h load_b_packed(const half_t* __restrict__ pk, int frag, int lane) {
  const half_t* p = pk + (((size_t)frag * 32 + lane) << 4);
  const v8h lo = *(const v8h*)p;
  const v8h hi = *(const v8h*)(p + 8);
  v16h b;
#pragma unroll
  for (int i = 0; i < 8; ++i) { b[i] = lo[i]; b[8 + i] = hi[i]; }
  return b;
}

// A fragment (16x32) from row-major fp32 global matrix, column = K*stride+off.
DEVINL v16h load_a_frag_g32(const float* __restrict__ X, int ld, int row0,
                            int k0, int stride, int off, int lane) {
  const int r  = row0 + (lane & 15);
  const int kb = (lane & 16) ? 8 : 0;
  const float* p = X + (size_t)r * ld;
  v16h a;
#pragma unroll
  for (int i = 0; i < 8; ++i) {
    const int K0 = k0 + kb + i;
    a[i]     = (half_t)p[K0 * stride + off];
    a[8 + i] = (half_t)p[(K0 + 16) * stride + off];
  }
  return a;
}

// A fragment from an LDS f16 tile (row-major): two contiguous 16-byte DS loads.
DEVINL v16h load_a_frag_lds(const half_t* h, int ld, int k0, int lane) {
  const int r  = lane & 15;
  const int kb = (lane & 16) ? 8 : 0;
  const half_t* p0 = h + r * ld + k0 + kb;
  const v8h lo = *(const v8h*)p0;
  const v8h hi = *(const v8h*)(p0 + 16);
  v16h a;
#pragma unroll
  for (int i = 0; i < 8; ++i) { a[i] = lo[i]; a[8 + i] = hi[i]; }
  return a;
}

// A fragment for edge_feats (16x8, zero-padded to 16x32).
DEVINL v16h load_a_frag_ef(const float* __restrict__ ef, int row0, int lane) {
  const int r  = row0 + (lane & 15);
  const int kb = (lane & 16) ? 8 : 0;
  v16h a;
#pragma unroll
  for (int i = 0; i < 8; ++i) {
    const int K = kb + i;
    a[i]     = (K < 8) ? (half_t)ef[r * 8 + K] : (half_t)0.0f;
    a[8 + i] = (half_t)0.0f;
  }
  return a;
}

DEVINL float silu(float x) { return x / (1.0f + __expf(-x)); }

// CDNA5 async global->LDS copy, 16B per lane, INST_OFFSET applies to both the
// LDS and global addresses (CDNA5 ISA).  Tracked by ASYNCcnt.
DEVINL void async_copy_b128(unsigned lds_addr, unsigned long long gaddr) {
  asm volatile("global_load_async_to_lds_b128 %0, %1, off"
               :: "v"(lds_addr), "v"(gaddr) : "memory");
}
DEVINL void async_copy_b128_off512(unsigned lds_addr, unsigned long long gaddr) {
  asm volatile("global_load_async_to_lds_b128 %0, %1, off offset:512"
               :: "v"(lds_addr), "v"(gaddr) : "memory");
}
DEVINL void wait_asynccnt0() {
  asm volatile("s_wait_asynccnt 0" ::: "memory");
}

// ===========================================================================
// Kernel 0: one-shot weight pack into f16 B-fragment layout.
// ===========================================================================
__global__ __launch_bounds__(32)
void pack_weights_kernel(const float* __restrict__ Wup0, const float* __restrict__ Wup1,
                         const float* __restrict__ mw1, const float* __restrict__ mw2,
                         const float* __restrict__ mw3, const float* __restrict__ mw4,
                         const float* __restrict__ Wl0, const float* __restrict__ Wl1,
                         half_t* __restrict__ pk) {
  const int lane = threadIdx.x;
  int f = 0;
  auto emit = [&](const float* W, int ldN, int k0, int n0, int kmax) {
    const v16h b = make_b_frag(W, ldN, k0, n0, kmax, lane);
    half_t* p = pk + (((size_t)f * 32 + lane) << 4);
    v8h lo, hi;
#pragma unroll
    for (int i = 0; i < 8; ++i) { lo[i] = b[i]; hi[i] = b[8 + i]; }
    *(v8h*)p = lo;
    *(v8h*)(p + 8) = hi;
    ++f;
  };
  for (int nt = 0; nt < 4; ++nt)  for (int kc = 0; kc < 2; ++kc) emit(Wup0, 64, kc * 32, nt * 16, 64);
  for (int nt = 0; nt < 4; ++nt)  for (int kc = 0; kc < 2; ++kc) emit(Wup1, 64, kc * 32, nt * 16, 64);
  for (int nt = 0; nt < 4; ++nt)                                 emit(mw1,  64, 0,       nt * 16, 8);
  for (int nt = 0; nt < 4; ++nt)  for (int kc = 0; kc < 2; ++kc) emit(mw2,  64, kc * 32, nt * 16, 64);
  for (int nt = 0; nt < 4; ++nt)  for (int kc = 0; kc < 2; ++kc) emit(mw3,  64, kc * 32, nt * 16, 64);
  for (int nt = 0; nt < 16; ++nt) for (int kc = 0; kc < 2; ++kc) emit(mw4, 256, kc * 32, nt * 16, 64);
  for (int nt = 0; nt < 4; ++nt)  for (int kc = 0; kc < 4; ++kc) emit(Wl0,  64, kc * 32, nt * 16, 128);
  for (int nt = 0; nt < 4; ++nt)  for (int kc = 0; kc < 4; ++kc) emit(Wl1,  64, kc * 32, nt * 16, 128);
}

// ===========================================================================
// Kernel 1: node up-projection into combined xcat[node][256] workspace:
// cols 0..63 = x0 = nf[:, :64] @ Wup0 ; cols 64+d*3+m = x1[n,d,m].
// One wave per 16-node tile; LDS-staged coalesced b128 stores.
// ===========================================================================
__global__ __launch_bounds__(32)
void node_up_kernel(const float* __restrict__ nf, const half_t* __restrict__ pk,
                    float* __restrict__ xcat) {
  __shared__ float st[16 * 256];
  const int lane = threadIdx.x;
  const int row0 = blockIdx.x * 16;

  v16h a0k[2], a1k[3][2];
#pragma unroll
  for (int kc = 0; kc < 2; ++kc) {
    a0k[kc] = load_a_frag_g32(nf, 256, row0, kc * 32, 1, 0, lane);
#pragma unroll
    for (int m = 0; m < 3; ++m)
      a1k[m][kc] = load_a_frag_g32(nf, 256, row0, kc * 32, 3, 64 + m, lane);
  }

  const int n  = lane & 15;
  const int mb = (lane & 16) ? 8 : 0;
#pragma unroll
  for (int nt = 0; nt < 4; ++nt) {
    const int n0 = nt * 16;
    v8f acc = {};
#pragma unroll
    for (int kc = 0; kc < 2; ++kc)
      acc = wmma_f16(a0k[kc], load_b_packed(pk, FRAG_UP0 + nt * 2 + kc, lane), acc);
#pragma unroll
    for (int v = 0; v < 8; ++v) st[(mb + v) * 256 + n0 + n] = acc[v];

    v16h b1[2];
#pragma unroll
    for (int kc = 0; kc < 2; ++kc) b1[kc] = load_b_packed(pk, FRAG_UP1 + nt * 2 + kc, lane);
#pragma unroll
    for (int m = 0; m < 3; ++m) {
      v8f am = {};
#pragma unroll
      for (int kc = 0; kc < 2; ++kc) am = wmma_f16(a1k[m][kc], b1[kc], am);
#pragma unroll
      for (int v = 0; v < 8; ++v)
        st[(mb + v) * 256 + 64 + (n0 + n) * 3 + m] = am[v];
    }
  }
  __syncthreads();
  // coalesced write: 16 rows x 1KB contiguous
  float4* dst = (float4*)(xcat + (size_t)row0 * 256);
  const float4* src = (const float4*)st;
#pragma unroll
  for (int it = 0; it < 32; ++it) dst[it * 32 + lane] = src[it * 32 + lane];
}

// ===========================================================================
// Kernel 2: fused edge pipeline, one wave per 16-edge tile.
// Async global->LDS gather overlapped with the MLP (WMMA+silu) -> layer4
// fused with tensor-product elementwise (consumed straight from D-fragments)
// -> mid0/mid1 f16 LDS -> final WMMA linears -> LDS-staged coalesced output.
// ===========================================================================
__global__ __launch_bounds__(32)
void edge_kernel(const float* __restrict__ ef, const float* __restrict__ ea,
                 const int* __restrict__ eidx, const float* __restrict__ xcat,
                 const half_t* __restrict__ pk, float* __restrict__ out) {
  // LDS carve-out (~33 KB):
  //  [0,16384)      s_ls : gathered xcat rows (f32 16x256); reused as out stage
  //  [16384,32768)  mid0 (f16 16x128) + mid1[3] (f16 16x128 each)
  //                 h_lds (f16 16x64, 2KB) aliases the start of mid0
  //  [32768,33024)  a_lds (f32 16x4)
  //  [33024,33088)  snd  (int 16)
  __shared__ __align__(16) unsigned char smem[33088];
  float*  s_ls  = (float*)smem;
  half_t* mid0  = (half_t*)(smem + 16384);
  half_t* mid1  = (half_t*)(smem + 16384 + 4096);  // [m*2048 + e*128 + c]
  half_t* h_lds = (half_t*)(smem + 16384);         // alias (dead before mid0 writes)
  float*  a_lds = (float*)(smem + 32768);
  int*    snd   = (int*)(smem + 33024);

  const int lane = threadIdx.x;
  const int e0   = blockIdx.x * 16;
  const int n    = lane & 15;
  const int mb   = (lane & 16) ? 8 : 0;

  // ---- senders + edge_attrs --------------------------------------------
  if (lane < 16) {
    snd[lane] = eidx[e0 + lane];  // edge_index row 0 = senders
    ((float4*)a_lds)[lane] = *(const float4*)(ea + (size_t)(e0 + lane) * 4);
  }
  __syncthreads();

  // ---- async gather: one contiguous 1KB row per edge, global -> LDS -----
  // Issued now, consumed after the MLP; ~20 WMMAs hide the random latency.
  {
    const unsigned lbase = (unsigned)(unsigned long long)(void*)s_ls;
    const unsigned long long gbase =
        (unsigned long long)(const void*)xcat + (unsigned long long)lane * 16u;
    for (int e = 0; e < 16; ++e) {
      const unsigned long long g = gbase + (unsigned long long)snd[e] * 1024u;
      const unsigned l = lbase + (unsigned)(e * 1024 + lane * 16);
      async_copy_b128(l, g);            // bytes [0,512)  of the row
      async_copy_b128_off512(l, g);     // bytes [512,1024)
    }
  }

  // ---- MLP layer 1: ef(16x8 padded) @ mw1 -> silu -> h ------------------
  {
    const v16h a = load_a_frag_ef(ef, e0, lane);
#pragma unroll
    for (int nt = 0; nt < 4; ++nt) {
      v8f acc = {};
      acc = wmma_f16(a, load_b_packed(pk, FRAG_MW1 + nt, lane), acc);
#pragma unroll
      for (int v = 0; v < 8; ++v)
        h_lds[(mb + v) * 64 + nt * 16 + n] = (half_t)silu(acc[v]);
    }
  }
  __syncthreads();

  // ---- MLP layers 2,3 ---------------------------------------------------
#pragma unroll
  for (int layer = 0; layer < 2; ++layer) {
    const int fb = (layer == 0) ? FRAG_MW2 : FRAG_MW3;
    const v16h a0 = load_a_frag_lds(h_lds, 64, 0, lane);
    const v16h a1 = load_a_frag_lds(h_lds, 64, 32, lane);
    __syncthreads();
#pragma unroll
    for (int nt = 0; nt < 4; ++nt) {
      v8f acc = {};
      acc = wmma_f16(a0, load_b_packed(pk, fb + nt * 2 + 0, lane), acc);
      acc = wmma_f16(a1, load_b_packed(pk, fb + nt * 2 + 1, lane), acc);
#pragma unroll
      for (int v = 0; v < 8; ++v)
        h_lds[(mb + v) * 64 + nt * 16 + n] = (half_t)silu(acc[v]);
    }
    __syncthreads();
  }

  // ---- layer 4 fused with tensor-product elementwise --------------------
  // tp_w tile nt: j = nt>>2 selects w1..w4, channel c = (nt&3)*16 + n;
  // acc[v] = w_j[edge mb+v][c].  mid0/mid1 built directly from D-fragments.
  {
    const v16h a0 = load_a_frag_lds(h_lds, 64, 0, lane);
    const v16h a1 = load_a_frag_lds(h_lds, 64, 32, lane);
    __syncthreads();     // h region (aliased by mid0) now dead
    wait_asynccnt0();    // gathered s_ls rows are now resident in LDS
#pragma unroll
    for (int j = 0; j < 4; ++j) {
#pragma unroll
      for (int t = 0; t < 4; ++t) {
        const int nt = j * 4 + t;
        v8f acc = {};
        acc = wmma_f16(a0, load_b_packed(pk, FRAG_MW4 + nt * 2 + 0, lane), acc);
        acc = wmma_f16(a1, load_b_packed(pk, FRAG_MW4 + nt * 2 + 1, lane), acc);
        const int c = t * 16 + n;
#pragma unroll
        for (int v = 0; v < 8; ++v) {
          const int e = mb + v;
          const float* sr = s_ls + e * 256;
          const float A0 = a_lds[e * 4 + 0];
          if (j == 0) {                       // o0a = w1*s0*a0
            mid0[e * 128 + c] = (half_t)(acc[v] * sr[c] * A0);
          } else if (j == 1) {                // o1a = w2*s0*a1m
            const float t2 = acc[v] * sr[c];
            mid1[0 * 2048 + e * 128 + c] = (half_t)(t2 * a_lds[e * 4 + 1]);
            mid1[1 * 2048 + e * 128 + c] = (half_t)(t2 * a_lds[e * 4 + 2]);
            mid1[2 * 2048 + e * 128 + c] = (half_t)(t2 * a_lds[e * 4 + 3]);
          } else if (j == 2) {                // o1b = w3*s1m*a0
            const float t3 = acc[v] * A0;
            mid1[0 * 2048 + e * 128 + 64 + c] = (half_t)(t3 * sr[64 + c * 3 + 0]);
            mid1[1 * 2048 + e * 128 + 64 + c] = (half_t)(t3 * sr[64 + c * 3 + 1]);
            mid1[2 * 2048 + e * 128 + 64 + c] = (half_t)(t3 * sr[64 + c * 3 + 2]);
          } else {                            // o0b = w4*dot(s1,a1)/sqrt3
            const float dot = (sr[64 + c * 3 + 0] * a_lds[e * 4 + 1] +
                               sr[64 + c * 3 + 1] * a_lds[e * 4 + 2] +
                               sr[64 + c * 3 + 2] * a_lds[e * 4 + 3]) * INV_SQRT3;
            mid0[e * 128 + 64 + c] = (half_t)(acc[v] * dot);
          }
        }
      }
    }
  }
  __syncthreads();

  // ---- final linears: y0 = mid0 @ Wl0 ; y1_m = mid1_m @ Wl1 -------------
  // Results staged into s_ls (out tile layout [e][col]) for coalesced store.
  {
    v16h af0[4];
#pragma unroll
    for (int kc = 0; kc < 4; ++kc) af0[kc] = load_a_frag_lds(mid0, 128, kc * 32, lane);
#pragma unroll
    for (int nt = 0; nt < 4; ++nt) {
      v8f acc = {};
#pragma unroll
      for (int kc = 0; kc < 4; ++kc)
        acc = wmma_f16(af0[kc], load_b_packed(pk, FRAG_WL0 + nt * 4 + kc, lane), acc);
#pragma unroll
      for (int v = 0; v < 8; ++v) s_ls[(mb + v) * 256 + nt * 16 + n] = acc[v];
    }
#pragma unroll
    for (int m = 0; m < 3; ++m) {
      v16h afm[4];
#pragma unroll
      for (int kc = 0; kc < 4; ++kc)
        afm[kc] = load_a_frag_lds(mid1 + m * 2048, 128, kc * 32, lane);
#pragma unroll
      for (int nt = 0; nt < 4; ++nt) {
        v8f acc = {};
#pragma unroll
        for (int kc = 0; kc < 4; ++kc)
          acc = wmma_f16(afm[kc], load_b_packed(pk, FRAG_WL1 + nt * 4 + kc, lane), acc);
#pragma unroll
        for (int v = 0; v < 8; ++v)
          s_ls[(mb + v) * 256 + 64 + (nt * 16 + n) * 3 + m] = acc[v];
      }
    }
  }
  __syncthreads();

  // ---- coalesced output: 16 KB contiguous b128 stores -------------------
  float4* dst = (float4*)(out + (size_t)e0 * 256);
  const float4* src = (const float4*)s_ls;
#pragma unroll
  for (int it = 0; it < 32; ++it) dst[it * 32 + lane] = src[it * 32 + lane];
}

// ===========================================================================
extern "C" void kernel_launch(void* const* d_in, const int* in_sizes, int n_in,
                              void* d_out, int out_size, void* d_ws, size_t ws_size,
                              hipStream_t stream) {
  (void)in_sizes; (void)n_in; (void)out_size; (void)ws_size;
  const float* node_feats = (const float*)d_in[0];
  const float* edge_attrs = (const float*)d_in[1];
  const float* edge_feats = (const float*)d_in[2];
  const float* W_up0      = (const float*)d_in[3];
  const float* W_up1      = (const float*)d_in[4];
  const float* mlp_w1     = (const float*)d_in[5];
  const float* mlp_w2     = (const float*)d_in[6];
  const float* mlp_w3     = (const float*)d_in[7];
  const float* mlp_w4     = (const float*)d_in[8];
  const float* W_l0       = (const float*)d_in[9];
  const float* W_l1       = (const float*)d_in[10];
  const int*   edge_index = (const int*)d_in[11];
  float* out = (float*)d_out;

  half_t* pk  = (half_t*)d_ws;                            // 100 KB packed weights
  float* xcat = (float*)((char*)d_ws + PK_BYTES);         // [N_NODES][256] fp32

  pack_weights_kernel<<<1, 32, 0, stream>>>(W_up0, W_up1, mlp_w1, mlp_w2,
                                            mlp_w3, mlp_w4, W_l0, W_l1, pk);
  node_up_kernel<<<N_NODES / 16, 32, 0, stream>>>(node_feats, pk, xcat);
  edge_kernel<<<N_EDGES / 16, 32, 0, stream>>>(edge_feats, edge_attrs, edge_index,
                                               xcat, pk, out);
}
